// MaskGenerator_65068754535090
// MI455X (gfx1250) — matrix operand order repaired and checked by reading, not compiled, not run on hardware
//
#include <hip/hip_runtime.h>
#include <hip/hip_bf16.h>
#include <math.h>

typedef __attribute__((ext_vector_type(16))) _Float16 v16h;
typedef __attribute__((ext_vector_type(8)))  float    v8f;

#define T_LEN 2048
#define B_SZ  16
#define C_IN  512
#define HID   256
#define G4H   1024   // 4*H
#define H2    512    // 2*H (BN channels)
#define NCHK  256    // BN n-chunks per direction (each 128 rows of T*B)

// ---- workspace layout (bytes) ----
#define OFF_XS   ((size_t)0)                                   // f16 [T][B][C]      32 MB
#define OFF_WIH  (OFF_XS   + (size_t)T_LEN*B_SZ*C_IN*2)        // f16 [2][4H][C]      2 MB
#define OFF_WHH  (OFF_WIH  + (size_t)2*G4H*C_IN*2)             // f16 [2][4H][H]      1 MB
#define OFF_BS   (OFF_WHH  + (size_t)2*G4H*HID*2)              // f32 [2][4H]         8 KB
#define OFF_PROJ (OFF_BS   + (size_t)2*G4H*4)                  // f32 [2][T*B][4H]  256 MB
#define OFF_HS   (OFF_PROJ + (size_t)2*T_LEN*B_SZ*G4H*4)       // f32 [2][T][B][H]   64 MB
#define OFF_SC   (OFF_HS   + (size_t)2*T_LEN*B_SZ*HID*4)       // f32 [512]
#define OFF_SH   (OFF_SC   + (size_t)H2*4)
#define OFF_M0   (OFF_SH   + (size_t)H2*4)                     // f32 [B][T]
#define OFF_M1   (OFF_M0   + (size_t)B_SZ*T_LEN*4)
#define OFF_PS   (OFF_M1   + (size_t)B_SZ*T_LEN*4)             // f32 [512][NCHK] partial sums
#define OFF_QS   (OFF_PS   + (size_t)H2*NCHK*4)                // f32 [512][NCHK] partial sumsq

__device__ __forceinline__ float sigf(float x){ return 1.0f/(1.0f+__expf(-x)); }

__device__ __forceinline__ v8f wmma_f16(v16h a, v16h b, v8f c){
  return __builtin_amdgcn_wmma_f32_16x16x32_f16(false, a, false, b, (short)0, c, false, false);
}

// ---------------- x transpose: [B][C][T] f32 -> [T][B][C] f16 (LDS tiled) ----------------
__global__ __launch_bounds__(256) void k_cvt_x(const float* __restrict__ x,
                                               _Float16* __restrict__ xs){
  __shared__ float tile[32][33];
  int t0 = blockIdx.x*32;           // T tile
  int c0 = blockIdx.y*32;           // C tile
  int b  = blockIdx.z;
  int tx = threadIdx.x & 31;
  int ty = threadIdx.x >> 5;        // 0..7
  const float* src = x + ((size_t)b*C_IN + c0)*T_LEN + t0;
  #pragma unroll
  for (int i = 0; i < 4; ++i){
    int c = ty + i*8;
    tile[c][tx] = src[(size_t)c*T_LEN + tx];     // coalesced along t
  }
  __syncthreads();
  #pragma unroll
  for (int i = 0; i < 4; ++i){
    int t = ty + i*8;
    xs[((size_t)(t0 + t)*B_SZ + b)*C_IN + c0 + tx] = (_Float16)tile[tx][t]; // coalesced along c
  }
}

__global__ __launch_bounds__(256) void k_prep_w(
    const float* __restrict__ wf, const float* __restrict__ wb,
    const float* __restrict__ hf, const float* __restrict__ hb,
    const float* __restrict__ bif, const float* __restrict__ bhf,
    const float* __restrict__ bib, const float* __restrict__ bhb,
    _Float16* __restrict__ wih, _Float16* __restrict__ whh, float* __restrict__ bs){
  int idx = blockIdx.x*256 + threadIdx.x;     // 2*4H*C = 1,048,576 exact
  {
    int dir = idx / (G4H*C_IN); int r = idx % (G4H*C_IN);
    wih[idx] = (_Float16)((dir ? wb : wf)[r]);
  }
  if (idx < 2*G4H*HID){
    int dir = idx / (G4H*HID); int r = idx % (G4H*HID);
    whh[idx] = (_Float16)((dir ? hb : hf)[r]);
  }
  if (idx < 2*G4H){
    int dir = idx / G4H; int r = idx % G4H;
    bs[idx] = dir ? (bib[r] + bhb[r]) : (bif[r] + bhf[r]);
  }
}

// ---------------- input projection GEMM ----------------
// proj[dir][m][n] = sum_c xs[m][c]*Wih[dir][n][c] + (bih+bhh)[dir][n]
__global__ __launch_bounds__(256) void k_proj(const _Float16* __restrict__ xs,
    const _Float16* __restrict__ wih, const float* __restrict__ bs,
    float* __restrict__ proj){
  int lane = threadIdx.x & 31;
  int wv   = threadIdx.x >> 5;
  int gw   = blockIdx.x*8 + wv;               // tile id < 2*2048*64
  int dir  = gw / (2048*64);
  int rem  = gw % (2048*64);
  int mt   = rem >> 6;
  int nt   = rem & 63;
  int grp  = lane >> 4, lr = lane & 15;
  const _Float16* arow = xs  + ((size_t)mt*16 + lr)*C_IN;
  const _Float16* brow = wih + ((size_t)dir*G4H + nt*16 + lr)*C_IN;
  v8f acc = {};
  for (int k = 0; k < 16; ++k){
    int kb = k*32;
    v16h a, b;
    #pragma unroll
    for (int q = 0; q < 8; ++q){
      a[q]   = arow[kb + grp*8 + q];
      a[q+8] = arow[kb + 16 + grp*8 + q];
    }
    #pragma unroll
    for (int q = 0; q < 16; ++q) b[q] = brow[kb + grp*8 + q + (q >= 8 ? 8 : 0)];
    acc = wmma_f16(a, b, acc);
  }
  float* pr = proj + (size_t)dir*T_LEN*B_SZ*G4H;
  #pragma unroll
  for (int k = 0; k < 8; ++k){
    int m = mt*16 + k + grp*8;
    int n = nt*16 + lr;
    pr[(size_t)m*G4H + n] = acc[k] + bs[dir*G4H + n];
  }
}

// ---------------- sequential LSTM recurrence ----------------
// one block per direction, 16 waves; wave w owns hidden cols j=w*16..w*16+15 and
// all four gate tiles for those columns. Whh B-fragments are register-resident
// across the whole T loop (4 gates x 8 ksteps x v16h = 256 VGPRs).
__global__ __launch_bounds__(512) void k_lstm(const _Float16* __restrict__ whh,
    const float* __restrict__ proj, float* __restrict__ hs){
  __shared__ _Float16 hl[B_SZ*HID];           // 8 KB f16 broadcast of h
  int dir  = blockIdx.x;
  int tid  = threadIdx.x;
  int lane = tid & 31, wv = tid >> 5;         // wv 0..15
  int grp  = lane >> 4, lr = lane & 15;
  int jb   = wv*16;
  for (int i = tid; i < B_SZ*HID; i += 512) hl[i] = (_Float16)0.f;
  __syncthreads();

  const _Float16* wbase = whh  + (size_t)dir*G4H*HID;
  const float*    pbase = proj + (size_t)dir*T_LEN*B_SZ*G4H;
  float*          hbase = hs   + (size_t)dir*T_LEN*B_SZ*HID;

  // ---- preload Whh fragments into registers (loop-invariant) ----
  v16h bw[4][8];
  #pragma unroll
  for (int g4 = 0; g4 < 4; ++g4){
    const _Float16* wrow = wbase + (size_t)(g4*256 + jb + lr)*HID + grp*8;
    #pragma unroll
    for (int ks = 0; ks < 8; ++ks){
      #pragma unroll
      for (int q = 0; q < 16; ++q) bw[g4][ks][q] = wrow[ks*32 + q + (q >= 8 ? 8 : 0)];
    }
  }

  float cst[8];
  #pragma unroll
  for (int k = 0; k < 8; ++k) cst[k] = 0.f;

  for (int t = 0; t < T_LEN; ++t){
    int time = dir ? (T_LEN - 1 - t) : t;
    // prefetch next step's 64KB proj slab (one 128B line per thread)
    if (t + 1 < T_LEN){
      int tn = dir ? (T_LEN - 2 - t) : (t + 1);
      __builtin_prefetch(pbase + (size_t)tn*B_SZ*G4H + tid*32, 0, 3);
    }
    v8f ai = {}, af = {}, ag = {}, ao = {};
    #pragma unroll
    for (int ks = 0; ks < 8; ++ks){
      int kb = ks*32;
      v16h a;
      #pragma unroll
      for (int q = 0; q < 8; ++q){
        a[q]   = hl[lr*HID + kb + grp*8 + q];
        a[q+8] = hl[lr*HID + kb + 16 + grp*8 + q];
      }
      ai = wmma_f16(a, bw[0][ks], ai);
      af = wmma_f16(a, bw[1][ks], af);
      ag = wmma_f16(a, bw[2][ks], ag);
      ao = wmma_f16(a, bw[3][ks], ao);
    }
    const float* pr = pbase + (size_t)time*B_SZ*G4H;
    float hv[8];
    #pragma unroll
    for (int k = 0; k < 8; ++k){
      int bb = k + grp*8;                     // batch row
      int j  = jb + lr;                       // hidden col
      float gi = ai[k] + pr[bb*G4H +       j];
      float gf = af[k] + pr[bb*G4H + 256 + j];
      float gg = ag[k] + pr[bb*G4H + 512 + j];
      float go = ao[k] + pr[bb*G4H + 768 + j];
      gi = sigf(gi); gf = sigf(gf); gg = tanhf(gg); go = sigf(go);
      cst[k] = gf*cst[k] + gi*gg;
      float h = go * tanhf(cst[k]);
      hv[k] = h;
      hbase[((size_t)time*B_SZ + bb)*HID + j] = h;
    }
    __syncthreads();                          // all waves done reading old h
    #pragma unroll
    for (int k = 0; k < 8; ++k)
      hl[(k + grp*8)*HID + jb + lr] = (_Float16)hv[k];
    __syncthreads();                          // new h visible to all waves
  }
}

// ---------------- BatchNorm: coalesced deterministic two-stage reduction ----------------
// stage 1: block = (dir, chunk of 128 rows of n=t*B+b); thread j reads hs[n][j] coalesced
__global__ __launch_bounds__(256) void k_bn_part(const float* __restrict__ hs,
    float* __restrict__ ps, float* __restrict__ qs){
  int dir   = blockIdx.x >> 8;
  int chunk = blockIdx.x & 255;               // 0..NCHK-1
  int j     = threadIdx.x;                    // 0..255
  const float* base = hs + (size_t)dir*T_LEN*B_SZ*HID;
  float s = 0.f, q = 0.f;
  int n0 = chunk*128;
  for (int n = n0; n < n0 + 128; ++n){
    float v = base[(size_t)n*HID + j];
    s += v; q += v*v;
  }
  int ch = dir*256 + j;
  ps[(size_t)ch*NCHK + chunk] = s;
  qs[(size_t)ch*NCHK + chunk] = q;
}
// stage 2: finalize fused scale/shift per channel
__global__ __launch_bounds__(256) void k_bn_fin(const float* __restrict__ ps,
    const float* __restrict__ qs, const float* __restrict__ gamma,
    const float* __restrict__ beta, float* __restrict__ sc, float* __restrict__ sh){
  int ch = blockIdx.x*256 + threadIdx.x;      // < 512
  float s = 0.f, q = 0.f;
  for (int c = 0; c < NCHK; ++c){
    s += ps[(size_t)ch*NCHK + c];
    q += qs[(size_t)ch*NCHK + c];
  }
  float mean = s * (1.f/(T_LEN*B_SZ));
  float var  = q * (1.f/(T_LEN*B_SZ)) - mean*mean;
  float g    = gamma[ch] * rsqrtf(var + 1e-5f);
  sc[ch] = g;
  sh[ch] = beta[ch] - mean*g;
}

// ---------------- linear + softmax + gumbel-ST -> posterior & sampled ----------------
__global__ __launch_bounds__(256) void k_post(const float* __restrict__ hs,
    const float* __restrict__ sc, const float* __restrict__ sh,
    const float* __restrict__ wlin, const float* __restrict__ blin,
    const float* __restrict__ u, const float* __restrict__ e,
    float* __restrict__ post, float* __restrict__ samp){
  int lane = threadIdx.x & 31, wv = threadIdx.x >> 5;
  int g = blockIdx.x*8 + wv;                  // = b*T + t, < 32768
  int b = g >> 11, t = g & 2047;
  float z0 = 0.f, z1 = 0.f;
  for (int ch = lane; ch < H2; ch += 32){
    int dir = ch >> 8, j = ch & 255;
    float h  = hs[(((size_t)dir*T_LEN + t)*B_SZ + b)*HID + j];
    float hn = h*sc[ch] + sh[ch];
    z0 += hn * wlin[ch];
    z1 += hn * wlin[H2 + ch];
  }
  #pragma unroll
  for (int o = 16; o > 0; o >>= 1){
    z0 += __shfl_xor(z0, o, 32);
    z1 += __shfl_xor(z1, o, 32);
  }
  if (lane == 0){
    z0 = (z0 + blin[0]) * 0.1f;               // /TEMP_SCALE
    z1 = (z1 + blin[1]) * 0.1f;
    float m  = fmaxf(z0, z1);
    float e0 = __expf(z0 - m), e1 = __expf(z1 - m);
    float inv = 1.f/(e0 + e1);
    float p0 = e0*inv, p1 = e1*inv;
    post[(size_t)g*2 + 0] = p0;
    post[(size_t)g*2 + 1] = p1;
    float u0 = u[(size_t)g*2 + 0], u1 = u[(size_t)g*2 + 1];
    float g0 = -__logf(-__logf(u0 + 1e-20f) + 1e-20f);
    float g1 = -__logf(-__logf(u1 + 1e-20f) + 1e-20f);
    float l0 = __logf(p0) + g0, l1 = __logf(p1) + g1;
    // straight-through value == one_hot(argmax); argmax over y == argmax over logits
    samp[g] = (l1 > l0) ? e[g] : 0.f;
  }
}

// ---------------- median pool k=5, reflect pad ----------------
__device__ __forceinline__ void cswap(float& a, float& b){
  float lo = fminf(a, b), hi = fmaxf(a, b); a = lo; b = hi;
}
__global__ __launch_bounds__(256) void k_med(const float* __restrict__ in,
                                             float* __restrict__ out){
  int g = blockIdx.x*256 + threadIdx.x;       // b*T + t
  int b = g >> 11, t = g & 2047;
  const float* row = in + (size_t)b*T_LEN;
  float v[5];
  #pragma unroll
  for (int k = 0; k < 5; ++k){
    int i = t + k - 2;
    if (i < 0) i = -i;
    if (i >= T_LEN) i = 2*T_LEN - 2 - i;
    v[k] = row[i];
  }
  cswap(v[0],v[1]); cswap(v[3],v[4]); cswap(v[2],v[4]); cswap(v[2],v[3]);
  cswap(v[0],v[3]); cswap(v[0],v[2]); cswap(v[1],v[4]); cswap(v[1],v[3]);
  cswap(v[1],v[2]);
  out[g] = v[2];
}

extern "C" void kernel_launch(void* const* d_in, const int* in_sizes, int n_in,
                              void* d_out, int out_size, void* d_ws, size_t ws_size,
                              hipStream_t stream){
  const float* x     = (const float*)d_in[0];
  const float* e     = (const float*)d_in[1];
  const float* u     = (const float*)d_in[2];
  const float* Wih_f = (const float*)d_in[3];
  const float* Whh_f = (const float*)d_in[4];
  const float* bih_f = (const float*)d_in[5];
  const float* bhh_f = (const float*)d_in[6];
  const float* Wih_b = (const float*)d_in[7];
  const float* Whh_b = (const float*)d_in[8];
  const float* bih_b = (const float*)d_in[9];
  const float* bhh_b = (const float*)d_in[10];
  const float* gamma = (const float*)d_in[11];
  const float* beta  = (const float*)d_in[12];
  const float* Wlin  = (const float*)d_in[13];
  const float* blin  = (const float*)d_in[14];

  char* ws = (char*)d_ws;
  _Float16* xs   = (_Float16*)(ws + OFF_XS);
  _Float16* wih  = (_Float16*)(ws + OFF_WIH);
  _Float16* whh  = (_Float16*)(ws + OFF_WHH);
  float*    bs   = (float*)(ws + OFF_BS);
  float*    proj = (float*)(ws + OFF_PROJ);
  float*    hsb  = (float*)(ws + OFF_HS);
  float*    sc   = (float*)(ws + OFF_SC);
  float*    sh   = (float*)(ws + OFF_SH);
  float*    m0   = (float*)(ws + OFF_M0);
  float*    m1   = (float*)(ws + OFF_M1);
  float*    ps   = (float*)(ws + OFF_PS);
  float*    qs   = (float*)(ws + OFF_QS);

  float* post     = (float*)d_out;                            // [B,T,2]
  float* mask_out = (float*)d_out + (size_t)B_SZ*T_LEN*2;     // [B,1,T]

  k_cvt_x <<<dim3(T_LEN/32, C_IN/32, B_SZ), 256, 0, stream>>>(x, xs);
  k_prep_w<<< 4096, 256, 0, stream>>>(Wih_f, Wih_b, Whh_f, Whh_b,
                                      bih_f, bhh_f, bih_b, bhh_b,
                                      wih, whh, bs);
  k_proj  <<<32768, 256, 0, stream>>>(xs, wih, bs, proj);
  k_lstm  <<<    2, 512, 0, stream>>>(whh, proj, hsb);
  k_bn_part<<< 512, 256, 0, stream>>>(hsb, ps, qs);
  k_bn_fin<<<    2, 256, 0, stream>>>(ps, qs, gamma, beta, sc, sh);
  k_post  <<< 4096, 256, 0, stream>>>(hsb, sc, sh, Wlin, blin, u, e, post, m0);
  k_med   <<<  128, 256, 0, stream>>>(m0, m1);
  k_med   <<<  128, 256, 0, stream>>>(m1, m0);
  k_med   <<<  128, 256, 0, stream>>>(m0, mask_out);
}
